// AE_73967926771956
// MI455X (gfx1250) — matrix operand order repaired and checked by reading, not compiled, not run on hardware
//
#include <hip/hip_runtime.h>
#include <cmath>

#ifndef __has_builtin
#define __has_builtin(x) 0
#endif

typedef __attribute__((ext_vector_type(2)))  float    v2f;
typedef __attribute__((ext_vector_type(8)))  float    v8f;
typedef __attribute__((ext_vector_type(16))) _Float16 v16h;

#if __has_builtin(__builtin_amdgcn_wmma_f32_16x16x4_f32)
#define USE_WMMA_F32X4 1
#else
#define USE_WMMA_F32X4 0
#endif

#define BM 32
#define BN 64
#define BK 32

__device__ __forceinline__ float actf(float x, int act) {
  if (act == 1) return x > 0.f ? x : 0.f;
  if (act == 2) return x > 0.f ? x : 0.2f * x;
  return x;
}

__device__ __forceinline__ void async_ld_b32(unsigned lds_addr, unsigned long long gaddr) {
  // CDNA5 async global->LDS copy, tracked by ASYNCcnt (CDNA5 ISA)
  asm volatile("global_load_async_to_lds_b32 %0, %1, off" :: "v"(lds_addr), "v"(gaddr) : "memory");
}

__device__ __forceinline__ void wait_async0() {
#if __has_builtin(__builtin_amdgcn_s_wait_asynccnt)
  __builtin_amdgcn_s_wait_asynccnt(0);
#else
  asm volatile("s_wait_asynccnt 0" ::: "memory");
#endif
}

struct GemmP {
  const float* Ap; const float* Bp; float* Cp; const float* bias;
  long long sA, sB, sC;
  int lda, ldb, ldc, M, N, K, tA, tB, act;
  float alpha;
};

// Generic batched GEMM: C = alpha * op(A) * op(B) (+ bias[m]) with activation.
// op(A)[m][k], op(B)[k][n]; tA/tB select transposed raw layouts.
// One wave computes a 16x16 f32 tile via V_WMMA_F32_16X16X4_F32 (8 steps/K-tile).
// Interior tiles are staged with global_load_async_to_lds_b32 (ASYNCcnt).
__launch_bounds__(256)
__global__ void k_gemm(GemmP p) {
  __shared__ float As[BM][BK + 4];  // M-major; row stride 36 floats (16B-mult, bank-spread)
  __shared__ float Bs[BK][BN + 4];  // K-major
  const int bz = blockIdx.z;
  const float* A  = p.Ap + p.sA * bz;
  const float* Bm = p.Bp + p.sB * bz;
  float*       Cp = p.Cp + p.sC * bz;
  const int m0 = blockIdx.y * BM;
  const int n0 = blockIdx.x * BN;
  const int tid  = threadIdx.x;
  const int lane = tid & 31;
  const int wv   = tid >> 5;
  const int wm   = (wv >> 2) * 16;
  const int wn   = (wv & 3) * 16;
  const int hi   = lane >> 4;
  const int l16  = lane & 15;
  v8f acc = {};
  for (int k0 = 0; k0 < p.K; k0 += BK) {
    if (k0 + BK < p.K) {  // CDNA5 global_prefetch_b8 of the next K tile
      __builtin_prefetch(A  + (long long)(m0 + (tid >> 3)) * p.lda + (k0 + BK), 0, 1);
      __builtin_prefetch(Bm + (long long)(k0 + BK + (tid >> 3)) * p.ldb + n0,   0, 1);
    }
    // ---- stage A tile (M-major in LDS): As[m][k] = opA(m0+m, k0+k)
    if (!p.tA) {
      int m  = tid >> 3;
      int kk = (tid & 7) * 4;
      const float* ap = A + (long long)(m0 + m) * p.lda + (k0 + kk);
      if ((m0 + BM <= p.M) && (k0 + BK <= p.K)) {
#pragma unroll
        for (int u = 0; u < 4; u++)
          async_ld_b32((unsigned)(size_t)&As[m][kk + u],
                       (unsigned long long)(size_t)(ap + u));
      } else {
#pragma unroll
        for (int u = 0; u < 4; u++) {
          float v = 0.f;
          if ((m0 + m) < p.M && (k0 + kk + u) < p.K) v = ap[u];
          As[m][kk + u] = v;
        }
      }
    } else {
      int kk = tid >> 3;
      int m  = (tid & 7) * 4;
      const float* ap = A + (long long)(k0 + kk) * p.lda + (m0 + m);
#pragma unroll
      for (int u = 0; u < 4; u++) {
        float v = 0.f;
        if ((k0 + kk) < p.K && (m0 + m + u) < p.M) v = ap[u];
        As[m + u][kk] = v;
      }
    }
    // ---- stage B tile: Bs[k][n] = opB(k0+k, n0+n)
    if (!p.tB) {
      int kk = tid >> 3;
      int n  = (tid & 7) * 8;
      const float* bp = Bm + (long long)(k0 + kk) * p.ldb + (n0 + n);
      if ((k0 + BK <= p.K) && (n0 + BN <= p.N)) {
#pragma unroll
        for (int u = 0; u < 8; u++)
          async_ld_b32((unsigned)(size_t)&Bs[kk][n + u],
                       (unsigned long long)(size_t)(bp + u));
      } else {
#pragma unroll
        for (int u = 0; u < 8; u++) {
          float v = 0.f;
          if ((k0 + kk) < p.K && (n0 + n + u) < p.N) v = bp[u];
          Bs[kk][n + u] = v;
        }
      }
    } else {
      int n  = tid >> 2;
      int kk = (tid & 3) * 8;
      const float* bp = Bm + (long long)(n0 + n) * p.ldb + (k0 + kk);
#pragma unroll
      for (int u = 0; u < 8; u++) {
        float v = 0.f;
        if ((n0 + n) < p.N && (k0 + kk + u) < p.K) v = bp[u];
        Bs[kk + u][n] = v;
      }
    }
    wait_async0();   // drain this wave's async copies (no-op if none issued)
    __syncthreads();
#if USE_WMMA_F32X4
#pragma unroll
    for (int kk = 0; kk < BK; kk += 4) {
      v2f a, b;
      // A 16x4 f32: lanes 0-15 -> K0,K1 ; lanes 16-31 -> K2,K3 ; M = lane&15
      a.x = As[wm + l16][kk + hi * 2 + 0];
      a.y = As[wm + l16][kk + hi * 2 + 1];
      // B 4x16 f32: VGPR0 K0/K2, VGPR1 K1/K3 across lane halves
      b.x = Bs[kk + hi * 2 + 0][wn + l16];
      b.y = Bs[kk + hi * 2 + 1][wn + l16];
      acc = __builtin_amdgcn_wmma_f32_16x16x4_f32(false, a, false, b, (short)0, acc,
                                                  false, false);
    }
#else
    {  // fallback: one 16x16x32 f16 WMMA per K tile (reduced precision)
      v16h ah, bh;
#pragma unroll
      for (int v = 0; v < 8; v++) {
        int ka = (v < 4 ? v * 2 : 16 + (v - 4) * 2) + hi * 8;
        ah[2 * v]     = (_Float16)As[wm + l16][ka];
        ah[2 * v + 1] = (_Float16)As[wm + l16][ka + 1];
        int kb = (v < 4 ? v * 2 : 16 + (v - 4) * 2) + hi * 8;
        bh[2 * v]     = (_Float16)Bs[kb][wn + l16];
        bh[2 * v + 1] = (_Float16)Bs[kb + 1][wn + l16];
      }
      acc = __builtin_amdgcn_wmma_f32_16x16x32_f16(false, ah, false, bh, (short)0, acc,
                                                   false, false);
    }
#endif
    __syncthreads();
  }
#pragma unroll
  for (int r = 0; r < 8; r++) {
    int m = m0 + wm + r + hi * 8;
    int n = n0 + wn + l16;
    if (m < p.M && n < p.N) {
      float v = p.alpha * acc[r];
      if (p.bias) v += p.bias[m];
      Cp[(long long)m * p.ldc + n] = actf(v, p.act);
    }
  }
}

// ---- BatchNorm1d over (B,C,N): per-channel stats ----
__global__ void k_bn3_stats(const float* x, float* mean, float* rstd, int B, int C, int N) {
  int c = blockIdx.x;
  __shared__ float s1[256], s2[256];
  float a = 0.f, b = 0.f;
  long long cn = (long long)C * N;
  long long tot = (long long)B * N;
  for (long long i = threadIdx.x; i < tot; i += 256) {
    long long bb = i / N, n = i % N;
    float v = x[bb * cn + (long long)c * N + n];
    a += v; b += v * v;
  }
  s1[threadIdx.x] = a; s2[threadIdx.x] = b; __syncthreads();
  for (int s = 128; s > 0; s >>= 1) {
    if (threadIdx.x < s) { s1[threadIdx.x] += s1[threadIdx.x + s]; s2[threadIdx.x] += s2[threadIdx.x + s]; }
    __syncthreads();
  }
  if (threadIdx.x == 0) {
    float m = s1[0] / (float)tot;
    float v = s2[0] / (float)tot - m * m;
    mean[c] = m; rstd[c] = rsqrtf(v + 1e-5f);
  }
}

__global__ void k_bn3_apply(float* x, const float* mean, const float* rstd,
                            int C, int N, long long total, int act) {
  for (long long i = blockIdx.x * 256LL + threadIdx.x; i < total; i += (long long)gridDim.x * 256) {
    int c = (int)((i / N) % C);
    x[i] = actf((x[i] - mean[c]) * rstd[c], act);
  }
}

// ---- BatchNorm1d over (C,B) layout, B small: fused per-row ----
__global__ void k_bn2_apply(float* x, int C, int B, int act) {
  int c = blockIdx.x * blockDim.x + threadIdx.x;
  if (c >= C) return;
  float* r = x + (long long)c * B;
  float m = 0.f;
  for (int b = 0; b < B; b++) m += r[b];
  m /= B;
  float v = 0.f;
  for (int b = 0; b < B; b++) { float d = r[b] - m; v += d * d; }
  v /= B;
  float rs = rsqrtf(v + 1e-5f);
  for (int b = 0; b < B; b++) r[b] = actf((r[b] - m) * rs, act);
}

// ---- max over N: (B,C,N) -> (C,B) ----
__global__ void k_rowmax(const float* x, float* out, int C, int N, int B) {
  int b = blockIdx.y, c = blockIdx.x;
  __shared__ float s[256];
  float m = -3.4e38f;
  const float* p = x + ((long long)b * C + c) * N;
  for (int n = threadIdx.x; n < N; n += 256) m = fmaxf(m, p[n]);
  s[threadIdx.x] = m; __syncthreads();
  for (int t = 128; t > 0; t >>= 1) {
    if (threadIdx.x < t) s[threadIdx.x] = fmaxf(s[threadIdx.x], s[threadIdx.x + t]);
    __syncthreads();
  }
  if (threadIdx.x == 0) out[(long long)c * B + b] = s[0];
}

__global__ void k_copy_chan(const float* src, float* dst, int B, int Cs, int N, int Cd, int coff) {
  long long total = (long long)B * Cs * N;
  for (long long i = blockIdx.x * 256LL + threadIdx.x; i < total; i += (long long)gridDim.x * 256) {
    long long n = i % N, c = (i / N) % Cs, b = i / ((long long)N * Cs);
    dst[(b * Cd + coff + c) * N + n] = src[i];
  }
}

__global__ void k_bcast_chan(const float* src, float* dst, int B, int Cs, int N, int Cd, int coff) {
  long long total = (long long)B * Cs * N;
  for (long long i = blockIdx.x * 256LL + threadIdx.x; i < total; i += (long long)gridDim.x * 256) {
    long long n = i % N, c = (i / N) % Cs, b = i / ((long long)N * Cs);
    dst[(b * Cd + coff + c) * N + n] = src[c * B + b];
  }
}

__global__ void k_softmax(float* x, int L) {
  __shared__ float sd[256];
  float* p = x + (long long)blockIdx.x * L;
  int t = threadIdx.x;
  float m = -3.4e38f;
  for (int i = t; i < L; i += 256) m = fmaxf(m, p[i]);
  sd[t] = m; __syncthreads();
  for (int s = 128; s > 0; s >>= 1) { if (t < s) sd[t] = fmaxf(sd[t], sd[t + s]); __syncthreads(); }
  m = sd[0]; __syncthreads();
  float sum = 0.f;
  for (int i = t; i < L; i += 256) { float e = expf(p[i] - m); p[i] = e; sum += e; }
  sd[t] = sum; __syncthreads();
  for (int s = 128; s > 0; s >>= 1) { if (t < s) sd[t] += sd[t + s]; __syncthreads(); }
  float inv = 1.f / sd[0];
  for (int i = t; i < L; i += 256) p[i] *= inv;
}

__global__ void k_gather(const float* x, const int* idx, float* out, int B, int C, int Nin, int M) {
  long long total = (long long)B * C * M;
  for (long long i = blockIdx.x * 256LL + threadIdx.x; i < total; i += (long long)gridDim.x * 256) {
    long long m = i % M, c = (i / M) % C, b = i / ((long long)M * C);
    out[i] = x[(b * C + c) * Nin + idx[b * M + m]];
  }
}

// Farthest point sampling. One block per batch. dist[] lives in LDS.
__global__ void k_fps(const float* x, int* cidx, int B, int C, int N, int ncent) {
  int b = blockIdx.x; (void)B;
  extern __shared__ float sm[];
  float* dist = sm;
  float* cent = sm + N;
  float* red  = cent + C;
  int*   redi = (int*)(red + 256);
  const float* xb = x + (long long)b * C * N;
  for (int n = threadIdx.x; n < N; n += 256) dist[n] = 1e10f;
  __syncthreads();
  int far = 0;
  for (int it = 0; it < ncent; it++) {
    if (threadIdx.x == 0) cidx[b * ncent + it] = far;
    for (int c = threadIdx.x; c < C; c += 256) cent[c] = xb[(long long)c * N + far];
    __syncthreads();
    for (int n = threadIdx.x; n < N; n += 256) {
      float d = 0.f;
      for (int c = 0; c < C; c++) { float df = xb[(long long)c * N + n] - cent[c]; d += df * df; }
      dist[n] = fminf(dist[n], d);
    }
    __syncthreads();
    float bm = -1.f; int bi = 0;
    for (int n = threadIdx.x; n < N; n += 256) { if (dist[n] > bm) { bm = dist[n]; bi = n; } }
    red[threadIdx.x] = bm; redi[threadIdx.x] = bi; __syncthreads();
    for (int t = 128; t > 0; t >>= 1) {
      if (threadIdx.x < t && red[threadIdx.x + t] > red[threadIdx.x]) {
        red[threadIdx.x] = red[threadIdx.x + t]; redi[threadIdx.x] = redi[threadIdx.x + t];
      }
      __syncthreads();
    }
    far = redi[0];
    __syncthreads();
  }
}

// GroupNorm + AdaGN affine fused. Block per (b,g). ss is (2C,B).
__global__ void k_adagn(const float* x, float* y, const float* ss, int B, int C, int N, int G) {
  int b = blockIdx.y, g = blockIdx.x;
  int cpg = C / G;
  long long sz = (long long)cpg * N;
  const float* xp = x + ((long long)b * C + (long long)g * cpg) * N;
  float*       yp = y + ((long long)b * C + (long long)g * cpg) * N;
  __shared__ float s1[256], s2[256], mv[2];
  float a = 0.f, q = 0.f;
  for (long long i = threadIdx.x; i < sz; i += 256) { float v = xp[i]; a += v; q += v * v; }
  s1[threadIdx.x] = a; s2[threadIdx.x] = q; __syncthreads();
  for (int s = 128; s > 0; s >>= 1) {
    if (threadIdx.x < s) { s1[threadIdx.x] += s1[threadIdx.x + s]; s2[threadIdx.x] += s2[threadIdx.x + s]; }
    __syncthreads();
  }
  if (threadIdx.x == 0) {
    float m = s1[0] / (float)sz;
    float v = s2[0] / (float)sz - m * m;
    mv[0] = m; mv[1] = rsqrtf(v + 1e-6f);
  }
  __syncthreads();
  float m = mv[0], rs = mv[1];
  for (long long i = threadIdx.x; i < sz; i += 256) {
    int c = g * cpg + (int)(i / N);
    float sc = ss[(long long)c * B + b];
    float sh = ss[(long long)(C + c) * B + b];
    yp[i] = (xp[i] - m) * rs * sc + sh;
  }
}

// 3-NN with inverse-distance weights: s (B,3,N), cents (B,3,M)
__global__ void k_topk3(const float* s, const float* cents, int* nidx, float* w,
                        int B, int N, int M) {
  long long i = blockIdx.x * 256LL + threadIdx.x;
  if (i >= (long long)B * N) return;
  int b = (int)(i / N), n = (int)(i % N);
  const float* sp = s + (long long)b * 3 * N;
  float px = sp[n], py = sp[N + n], pz = sp[2 * N + n];
  const float* cp = cents + (long long)b * 3 * M;
  float d0 = 3.4e38f, d1 = 3.4e38f, d2 = 3.4e38f;
  int i0 = 0, i1 = 0, i2 = 0;
  for (int m = 0; m < M; m++) {
    float dx = px - cp[m], dy = py - cp[M + m], dz = pz - cp[2 * M + m];
    float d = dx * dx + dy * dy + dz * dz;
    if (d < d0)      { d2 = d1; i2 = i1; d1 = d0; i1 = i0; d0 = d; i0 = m; }
    else if (d < d1) { d2 = d1; i2 = i1; d1 = d;  i1 = m; }
    else if (d < d2) { d2 = d;  i2 = m; }
  }
  float w0 = 1.f / (d0 + 1e-8f), w1 = 1.f / (d1 + 1e-8f), w2 = 1.f / (d2 + 1e-8f);
  float ws = w0 + w1 + w2;
  long long o = i * 3;
  nidx[o] = i0; nidx[o + 1] = i1; nidx[o + 2] = i2;
  w[o] = w0 / ws; w[o + 1] = w1 / ws; w[o + 2] = w2 / ws;
}

__global__ void k_interp(const float* feat, const int* ni, const float* w, float* out,
                         int B, int C, int N, int M) {
  long long total = (long long)B * C * N;
  for (long long i = blockIdx.x * 256LL + threadIdx.x; i < total; i += (long long)gridDim.x * 256) {
    long long n = i % N, c = (i / N) % C, b = i / ((long long)N * C);
    long long o = (b * N + n) * 3;
    const float* fp = feat + (b * C + c) * M;
    out[i] = w[o] * fp[ni[o]] + w[o + 1] * fp[ni[o + 1]] + w[o + 2] * fp[ni[o + 2]];
  }
}

__global__ void k_add(const float* a, const float* b, float* out, long long total) {
  for (long long i = blockIdx.x * 256LL + threadIdx.x; i < total; i += (long long)gridDim.x * 256)
    out[i] = a[i] + b[i];
}

// (C*C,B) -> (B,C,C) with +I
__global__ void k_taddeye(const float* tin, float* tout, int B, int C) {
  long long total = (long long)B * C * C;
  for (long long i = blockIdx.x * 256LL + threadIdx.x; i < total; i += (long long)gridDim.x * 256) {
    long long e = i % C, c = (i / C) % C, b = i / ((long long)C * C);
    float v = tin[(c * C + e) * B + b];
    if (c == e) v += 1.f;
    tout[i] = v;
  }
}

__global__ void k_reg(const float* tt, float* out, int B, int C) {
  __shared__ float s[256];
  float a = 0.f;
  long long total = (long long)B * C * C;
  for (long long i = threadIdx.x; i < total; i += 256) {
    int e = (int)(i % C), c = (int)((i / C) % C);
    float d = tt[i] - (c == e ? 1.f : 0.f);
    a += d * d;
  }
  s[threadIdx.x] = a; __syncthreads();
  for (int t = 128; t > 0; t >>= 1) { if (threadIdx.x < t) s[threadIdx.x] += s[threadIdx.x + t]; __syncthreads(); }
  if (threadIdx.x == 0) out[0] = 0.001f * s[0] / (float)B;
}

static inline int nblk(long long tot) {
  long long b = (tot + 255) / 256;
  if (b < 1) b = 1;
  if (b > 16384) b = 16384;
  return (int)b;
}

extern "C" void kernel_launch(void* const* d_in, const int* in_sizes, int n_in,
                              void* d_out, int out_size, void* d_ws, size_t ws_size,
                              hipStream_t stream) {
  (void)in_sizes; (void)n_in; (void)out_size; (void)ws_size;
  const int B = 8, N = 2048, CTX = 640;
  const float* x    = (const float*)d_in[0];
  const float* xt   = (const float*)d_in[1];
  const float* temb = (const float*)d_in[2];
  float* dout = (float*)d_out;

  // ---- arena ----
  char* arena = (char*)d_ws;
  size_t off = 0;
  auto af = [&](size_t n) -> float* {
    float* p = (float*)(arena + off);
    off += ((n * 4 + 255) / 256) * 256;
    return p;
  };
  auto ai = [&](size_t n) -> int* {
    int* p = (int*)(arena + off);
    off += ((n * 4 + 255) / 256) * 256;
    return p;
  };

  // ---- parameter decode (sorted-key pytree order; conv/attn dicts sorted) ----
  int pi = 3;
  struct ConvP { const float *b, *w; };
  struct AtP   { const float *kw, *ow, *qw, *vw; };
  struct AdaP  { const float *b, *w; };
  struct DnP   { AtP at; ConvP m0, m1; };
  struct TnP   { ConvP c1[3]; ConvP l2[3]; };
  auto TKf  = [&]() { return (const float*)d_in[pi++]; };
  auto tConv = [&]() { ConvP c; c.b = TKf(); c.w = TKf(); return c; };
  auto tAttn = [&]() { AtP a; a.kw = TKf(); a.ow = TKf(); a.qw = TKf(); a.vw = TKf(); return a; };
  auto tAda  = [&]() { AdaP a; a.b = TKf(); a.w = TKf(); return a; };
  auto tDn   = [&]() { DnP d; d.at = tAttn(); d.m0 = tConv(); d.m1 = tConv(); return d; };
  auto tTn   = [&]() {
    TnP t;
    t.c1[0] = tConv(); t.c1[1] = tConv(); t.c1[2] = tConv();
    t.l2[0] = tConv(); t.l2[1] = tConv(); t.l2[2] = tConv();
    return t;
  };
  AdaP an1 = tAda(); AdaP an2 = tAda(); AdaP an3 = tAda();
  AdaP an4 = tAda(); AdaP an5 = tAda(); AdaP an6 = tAda();
  DnP dn1 = tDn(); DnP dn2 = tDn(); DnP dn3 = tDn();
  AtP attn1 = tAttn(); AtP attn2 = tAttn();
  ConvP head0 = tConv(); ConvP head1 = tConv(); ConvP head2 = tConv();
  ConvP m11a = tConv(); ConvP m11b = tConv();
  ConvP m12a = tConv(); ConvP m12b = tConv();
  ConvP em2a = tConv(); ConvP em2b = tConv(); ConvP em2c = tConv();
  TnP tn3p = tTn(); TnP tn64p = tTn();
  ConvP initc = tConv();
  ConvP lin3 = tConv();
  ConvP dm0 = tConv(); ConvP dm1 = tConv(); ConvP dm2 = tConv();
  ConvP sup0 = tConv(); ConvP sup1 = tConv();
  ConvP te0 = tConv(); ConvP te1 = tConv();
  DnP up1 = tDn(); DnP up2 = tDn(); DnP up3 = tDn();

  // ---- long-lived buffers ----
  float* stat = af(4096);      // bn3 mean @0, rstd @2048
  float* z    = af(512 * B);   // (512,B)
  float* ctx  = af(CTX * B);   // (640,B)

  // ---- primitive launchers ----
  auto gemm = [&](const float* A, int lda, long long sA, int tA,
                  const float* Bp, int ldb, long long sB, int tB,
                  float* Cp, int ldc, long long sC,
                  int M, int Nn, int K, int nb, const float* bias, float alpha, int act) {
    GemmP p;
    p.Ap = A; p.Bp = Bp; p.Cp = Cp; p.bias = bias;
    p.sA = sA; p.sB = sB; p.sC = sC;
    p.lda = lda; p.ldb = ldb; p.ldc = ldc;
    p.M = M; p.N = Nn; p.K = K; p.tA = tA; p.tB = tB; p.act = act; p.alpha = alpha;
    dim3 g((Nn + BN - 1) / BN, (M + BM - 1) / BM, nb);
    k_gemm<<<g, 256, 0, stream>>>(p);
  };
  auto conv = [&](const ConvP& c, int Cout, int Cin, const float* X, float* Y, int Np, int act) {
    gemm(c.w, Cin, 0, 0, X, Np, (long long)Cin * Np, 0, Y, Np, (long long)Cout * Np,
         Cout, Np, Cin, B, c.b, 1.f, act);
  };
  auto bn3 = [&](float* X, int C, int Np, int act) {
    k_bn3_stats<<<C, 256, 0, stream>>>(X, stat, stat + 2048, B, C, Np);
    long long tot = (long long)B * C * Np;
    k_bn3_apply<<<nblk(tot), 256, 0, stream>>>(X, stat, stat + 2048, C, Np, tot, act);
  };
  auto copy_chan = [&](const float* src, float* dst, int Cs, int Np, int Cd, int coff) {
    k_copy_chan<<<nblk((long long)B * Cs * Np), 256, 0, stream>>>(src, dst, B, Cs, Np, Cd, coff);
  };
  auto attention = [&](const AtP& a, int qc, int kc, int D, int H,
                       const float* xq, int Nx, const float* y, int Ny, float* outp) {
    size_t mk = off;
    int HD = D * H;
    float* q  = af((size_t)B * HD * Nx);
    float* kk = af((size_t)B * HD * Ny);
    float* v  = af((size_t)B * HD * Ny);
    float* o  = af((size_t)B * HD * Nx);
    float* lg = af((size_t)Nx * Ny);
    gemm(a.qw, qc, 0, 0, xq, Nx, (long long)qc * Nx, 0, q, Nx, (long long)HD * Nx, HD, Nx, qc, B, nullptr, 1.f, 0);
    gemm(a.kw, kc, 0, 0, y,  Ny, (long long)kc * Ny, 0, kk, Ny, (long long)HD * Ny, HD, Ny, kc, B, nullptr, 1.f, 0);
    gemm(a.vw, kc, 0, 0, y,  Ny, (long long)kc * Ny, 0, v,  Ny, (long long)HD * Ny, HD, Ny, kc, B, nullptr, 1.f, 0);
    float sc = 1.f / sqrtf((float)D);
    for (int b = 0; b < B; b++)
      for (int h = 0; h < H; h++) {
        long long so = ((long long)b * H + h) * (long long)D;
        const float* qs = q + so * Nx;
        const float* ks = kk + so * Ny;
        const float* vs = v + so * Ny;
        float* os = o + so * Nx;
        gemm(qs, Nx, 0, 1, ks, Ny, 0, 0, lg, Ny, 0, Nx, Ny, D, 1, nullptr, sc, 0);
        k_softmax<<<Nx, 256, 0, stream>>>(lg, Ny);
        gemm(vs, Ny, 0, 0, lg, Ny, 0, 1, os, Nx, 0, D, Nx, Ny, 1, nullptr, 1.f, 0);
      }
    gemm(a.ow, HD, 0, 0, o, Nx, (long long)HD * Nx, 0, outp, Nx, (long long)D * Nx, D, Nx, HD, B, nullptr, 1.f, 0);
    off = mk;
  };
  auto tnet = [&](const TnP& t, int C, const float* xin, float* xo, int Np, float* tt) {
    size_t mk = off;
    float* ha = af((size_t)B * 1024 * Np);
    float* hb = af((size_t)B * 1024 * Np);
    conv(t.c1[0], 64, C, xin, ha, Np, 0);    bn3(ha, 64, Np, 1);
    conv(t.c1[1], 128, 64, ha, hb, Np, 0);   bn3(hb, 128, Np, 1);
    conv(t.c1[2], 1024, 128, hb, ha, Np, 0); bn3(ha, 1024, Np, 1);
    float* t0 = af(1024 * B);
    k_rowmax<<<dim3(1024, B), 256, 0, stream>>>(ha, t0, 1024, Np, B);
    float* t1 = af(512 * B);
    gemm(t.l2[0].w, 1024, 0, 0, t0, B, 0, 0, t1, B, 0, 512, B, 1024, 1, t.l2[0].b, 1.f, 0);
    k_bn2_apply<<<2, 256, 0, stream>>>(t1, 512, B, 1);
    float* t2 = af(256 * B);
    gemm(t.l2[1].w, 512, 0, 0, t1, B, 0, 0, t2, B, 0, 256, B, 512, 1, t.l2[1].b, 1.f, 0);
    k_bn2_apply<<<1, 256, 0, stream>>>(t2, 256, B, 1);
    float* t3 = af((size_t)C * C * B);
    gemm(t.l2[2].w, 256, 0, 0, t2, B, 0, 0, t3, B, 0, C * C, B, 256, 1, t.l2[2].b, 1.f, 0);
    float* tm = af((size_t)B * C * C);
    k_taddeye<<<nblk((long long)B * C * C), 256, 0, stream>>>(t3, tm, B, C);
    // xo[b] = tm[b]^T @ x[b]
    gemm(tm, C, (long long)C * C, 1, xin, Np, (long long)C * Np, 0,
         xo, Np, (long long)C * Np, C, Np, C, B, nullptr, 1.f, 0);
    if (tt)
      gemm(tm, C, (long long)C * C, 0, tm, C, (long long)C * C, 1,
           tt, C, (long long)C * C, C, C, C, B, nullptr, 1.f, 0);
    off = mk;
  };

  // ================= ENCODER =================
  {
    size_t mk = off;
    float* xo = af((size_t)B * 3 * N);
    tnet(tn3p, 3, x, xo, N, nullptr);
    float* ha = af((size_t)B * 64 * N);
    conv(m11a, 64, 3, xo, ha, N, 0); bn3(ha, 64, N, 1);
    float* x1 = af((size_t)B * 64 * N);
    conv(m11b, 64, 64, ha, x1, N, 0);
    float* x1t = af((size_t)B * 64 * N);
    float* tt = af((size_t)B * 64 * 64);
    tnet(tn64p, 64, x1, x1t, N, tt);
    k_reg<<<1, 256, 0, stream>>>(tt, dout + 49152, B, 64);
    float* xa = af((size_t)B * 128 * N);
    attention(attn1, 64, 64, 128, 2, x1t, N, x1t, N, xa);
    float* h2a = af((size_t)B * 256 * N);
    conv(m12a, 256, 128, xa, h2a, N, 0); bn3(h2a, 256, N, 1);
    float* h2b = af((size_t)B * 512 * N);
    conv(m12b, 512, 256, h2a, h2b, N, 0);
    float* g = af(512 * B);
    k_rowmax<<<dim3(512, B), 256, 0, stream>>>(h2b, g, 512, N, B);
    float* hc = af((size_t)B * 515 * N);
    k_bcast_chan<<<nblk((long long)B * 512 * N), 256, 0, stream>>>(g, hc, B, 512, N, 515, 0);
    copy_chan(xo, hc, 3, N, 515, 512);
    float* hA = af((size_t)B * 512 * N);
    conv(em2a, 512, 515, hc, hA, N, 0); bn3(hA, 512, N, 1);
    float* hB = af((size_t)B * 512 * N);
    conv(em2b, 512, 512, hA, hB, N, 0); bn3(hB, 512, N, 1);
    conv(em2c, 512, 512, hB, hA, N, 0);
    float* hT = af((size_t)B * 512 * N);
    attention(attn2, 512, 512, 512, 1, hA, N, hA, N, hT);
    float* hv = af(512 * B);
    k_rowmax<<<dim3(512, B), 256, 0, stream>>>(hT, hv, 512, N, B);
    float* u1 = af(256 * B);
    gemm(head0.w, 512, 0, 0, hv, B, 0, 0, u1, B, 0, 256, B, 512, 1, head0.b, 1.f, 0);
    k_bn2_apply<<<1, 256, 0, stream>>>(u1, 256, B, 0);
    float* u2 = af(256 * B);
    gemm(head1.w, 256, 0, 0, u1, B, 0, 0, u2, B, 0, 256, B, 256, 1, head1.b, 1.f, 0);
    k_bn2_apply<<<1, 256, 0, stream>>>(u2, 256, B, 0);
    gemm(head2.w, 256, 0, 0, u2, B, 0, 0, z, B, 0, 512, B, 256, 1, head2.b, 1.f, 0);
    off = mk;
  }

  // ================= DECODER =================
  {  // time embedding + ctx = concat(t, z) as (640,B)
    size_t mk = off;
    float* ta = af(256 * B);
    gemm(te0.w, 128, 0, 0, temb, 128, 0, 1, ta, B, 0, 256, B, 128, 1, te0.b, 1.f, 0);
    k_bn2_apply<<<1, 256, 0, stream>>>(ta, 256, B, 2);
    float* tb = af(128 * B);
    gemm(te1.w, 256, 0, 0, ta, B, 0, 0, tb, B, 0, 128, B, 256, 1, te1.b, 1.f, 0);
    hipMemcpyAsync(ctx, tb, (size_t)128 * B * 4, hipMemcpyDeviceToDevice, stream);
    hipMemcpyAsync(ctx + 128 * B, z, (size_t)512 * B * 4, hipMemcpyDeviceToDevice, stream);
    off = mk;
  }

  auto adagn = [&](const AdaP& a, float* xf, int C, int Np, int G) {
    size_t mk = off;
    float* ss = af((size_t)2 * C * B);
    gemm(a.w, CTX, 0, 0, ctx, B, 0, 0, ss, B, 0, 2 * C, B, CTX, 1, a.b, 1.f, 0);
    k_adagn<<<dim3(G, B), 256, 0, stream>>>(xf, xf, ss, B, C, Np, G);
    off = mk;
  };
  auto down = [&](const DnP& p, int ncent, int D, const float* xf, int Cf, int Np,
                  const int* idxp, float* fout, int* cidx) {
    size_t mk = off;
    int C = Cf + 3;
    float* xcat = af((size_t)B * C * Np);
    const float* xtc = xt;
    if (idxp) {
      float* xg = af((size_t)B * 3 * Np);
      k_gather<<<nblk((long long)B * 3 * Np), 256, 0, stream>>>(xt, idxp, xg, B, 3, N, Np);
      xtc = xg;
    }
    copy_chan(xf, xcat, Cf, Np, C, 0);
    copy_chan(xtc, xcat, 3, Np, C, Cf);
    size_t sh = ((size_t)Np + C + 256) * 4 + 256 * 4;
    k_fps<<<B, 256, sh, stream>>>(xcat, cidx, B, C, Np, ncent);
    float* cents = af((size_t)B * C * ncent);
    k_gather<<<nblk((long long)B * C * ncent), 256, 0, stream>>>(xcat, cidx, cents, B, C, Np, ncent);
    float* at = af((size_t)B * D * ncent);
    attention(p.at, C, C, D, 1, cents, ncent, xcat, Np, at);
    float* hh = af((size_t)B * D * ncent);
    conv(p.m0, D, D, at, hh, ncent, 0); bn3(hh, D, ncent, 2);
    conv(p.m1, D, D, hh, fout, ncent, 0);
    off = mk;
  };
  auto upsamp = [&](const DnP& p, const float* q, int Cq, int Nq,
                    const float* kk, int Ck, int Nk, int keyD, float* outb) {
    size_t mk = off;
    float* u = af((size_t)B * keyD * Nq);
    attention(p.at, Cq, Ck, keyD, 1, q, Nq, kk, Nk, u);
    int Cc = keyD + Cq;
    float* cat = af((size_t)B * Cc * Nq);
    copy_chan(u, cat, keyD, Nq, Cc, 0);
    copy_chan(q, cat, Cq, Nq, Cc, keyD);
    float* hh = af((size_t)B * Cq * Nq);
    conv(p.m0, Cq, Cc, cat, hh, Nq, 0); bn3(hh, Cq, Nq, 2);
    conv(p.m1, Cq, Cq, hh, outb, Nq, 0);
    off = mk;
  };
  auto interp = [&](const float* s, const float* feat, int C, int M, const int* idx, float* outb) {
    size_t mk = off;
    float* cents = af((size_t)B * 3 * M);
    k_gather<<<nblk((long long)B * 3 * M), 256, 0, stream>>>(s, idx, cents, B, 3, N, M);
    int* ni = ai((size_t)B * N * 3);
    float* w = af((size_t)B * N * 3);
    k_topk3<<<(B * N + 255) / 256, 256, 0, stream>>>(s, cents, ni, w, B, N, M);
    k_interp<<<nblk((long long)B * C * N), 256, 0, stream>>>(feat, ni, w, outb, B, C, N, M);
    off = mk;
  };

  float* f0 = af((size_t)B * 64 * N);
  {
    size_t mk = off;
    float* xin = af((size_t)B * 643 * N);
    copy_chan(xt, xin, 3, N, 643, 0);
    k_bcast_chan<<<nblk((long long)B * 640 * N), 256, 0, stream>>>(ctx, xin, B, 640, N, 643, 3);
    conv(initc, 64, 643, xin, f0, N, 0);
    off = mk;
  }
  int* idx1 = ai(B * 512); int* idx2 = ai(B * 128); int* idx3 = ai(B * 16);
  float* f1 = af((size_t)B * 128 * 512);
  down(dn1, 512, 128, f0, 64, N, nullptr, f1, idx1);
  adagn(an1, f1, 128, 512, 16);
  float* f2 = af((size_t)B * 256 * 128);
  down(dn2, 128, 256, f1, 128, 512, idx1, f2, idx2);
  adagn(an2, f2, 256, 128, 32);
  float* f3 = af((size_t)B * 512 * 16);
  down(dn3, 16, 512, f2, 256, 128, idx2, f3, idx3);
  adagn(an3, f3, 512, 16, 32);
  float* f2u = af((size_t)B * 256 * 128);
  upsamp(up1, f2, 256, 128, f3, 512, 16, 512, f2u);
  adagn(an4, f2u, 256, 128, 32);
  float* f1u = af((size_t)B * 128 * 512);
  upsamp(up2, f1, 128, 512, f2u, 256, 128, 256, f1u);
  adagn(an5, f1u, 128, 512, 16);
  float* f0u = af((size_t)B * 64 * N);
  upsamp(up3, f0, 64, N, f1u, 128, 512, 128, f0u);
  adagn(an6, f0u, 64, N, 8);
  // support net (shared across interpolations)
  float* s = af((size_t)B * 3 * N);
  {
    size_t mk = off;
    float* sh1 = af((size_t)B * 64 * N);
    conv(sup0, 64, 3, xt, sh1, N, 0); bn3(sh1, 64, N, 2);
    conv(sup1, 3, 64, sh1, s, N, 0);
    off = mk;
  }
  float* feat = af((size_t)B * 960 * N);
  copy_chan(f0u, feat, 64, N, 960, 0);
  {
    size_t mk = off;
    float* fi = af((size_t)B * 512 * N);
    interp(s, f1u, 128, 512, idx1, fi); copy_chan(fi, feat, 128, N, 960, 64);
    interp(s, f2u, 256, 128, idx2, fi); copy_chan(fi, feat, 256, N, 960, 192);
    interp(s, f3, 512, 16, idx3, fi);   copy_chan(fi, feat, 512, N, 960, 448);
    off = mk;
  }
  {
    size_t mk = off;
    float* hA = af((size_t)B * 1024 * N);
    conv(dm0, 1024, 960, feat, hA, N, 0); bn3(hA, 1024, N, 2);
    float* hB = af((size_t)B * 256 * N);
    conv(dm1, 256, 1024, hA, hB, N, 0); bn3(hB, 256, N, 2);
    float* o1 = af((size_t)B * 3 * N);
    conv(dm2, 3, 256, hB, o1, N, 0);
    float* o2 = af((size_t)B * 3 * N);
    conv(lin3, 3, 3, xt, o2, N, 0);
    k_add<<<nblk(49152), 256, 0, stream>>>(o1, o2, dout, 49152);
    off = mk;
  }
}
